// Order1GraphAttentionLayer_4784593567904
// MI455X (gfx1250) — matrix operand order repaired and checked by reading, not compiled, not run on hardware
//
#include <hip/hip_runtime.h>
#include <hip/hip_bf16.h>
#include <math.h>

typedef __attribute__((ext_vector_type(16))) _Float16 v16h;
typedef __attribute__((ext_vector_type(8)))  float    v8f;

#define GAT_B 4
#define GAT_N 4096
#define GAT_D 256
#define GAT_ALPHA 0.2f
#define GAT_NEG_INF -9e15f

// ---------- kernel 1: wa1 = W @ a1, wa2 = W @ a2 (tiny) ----------
__global__ void k_wa(const float* __restrict__ W, const float* __restrict__ a1,
                     const float* __restrict__ a2, float* __restrict__ wa1,
                     float* __restrict__ wa2) {
  int k = threadIdx.x;                       // 256 threads, 1 block
  const float* row = W + (size_t)k * GAT_D;
  float s1 = 0.f, s2 = 0.f;
  for (int d = 0; d < GAT_D; ++d) { float w = row[d]; s1 += w * a1[d]; s2 += w * a2[d]; }
  wa1[k] = s1; wa2[k] = s2;
}

// ---------- kernel 2: Ax/Ay = feat @ wa1/wa2, one wave per row ----------
__global__ void k_axay(const float* __restrict__ feat, const float* __restrict__ wa1,
                       const float* __restrict__ wa2, float* __restrict__ Ax,
                       float* __restrict__ Ay) {
  __shared__ float s1[GAT_D], s2[GAT_D];
  for (int i = threadIdx.x; i < GAT_D; i += 256) { s1[i] = wa1[i]; s2[i] = wa2[i]; }
  __syncthreads();
  int wave = threadIdx.x >> 5, lane = threadIdx.x & 31;
  int row = blockIdx.x * 8 + wave;           // row < B*N
  const float* fr = feat + (size_t)row * GAT_D;
  float a = 0.f, b = 0.f;
  for (int k = lane; k < GAT_D; k += 32) { float f = fr[k]; a += f * s1[k]; b += f * s2[k]; }
  #pragma unroll
  for (int m = 16; m >= 1; m >>= 1) { a += __shfl_xor(a, m, 32); b += __shfl_xor(b, m, 32); }
  if (lane == 0) { Ax[row] = a; Ay[row] = b; }
}

// ---------- kernel 3: per-row softmax stats (max, sum) ----------
__global__ void __launch_bounds__(256) k_stats(const int* __restrict__ adj,
                                               const float* __restrict__ Ax,
                                               const float* __restrict__ Ay,
                                               float* __restrict__ Mm,
                                               float* __restrict__ Ss) {
  const int rowG = blockIdx.x;               // b*N + i
  const int b = rowG / GAT_N;
  const int* arow = adj + (size_t)rowG * GAT_N;
  const float* AxB = Ax + (size_t)b * GAT_N;
  const float ay = Ay[rowG];
  float ev[GAT_N / 256];
  float lmax = GAT_NEG_INF;
  #pragma unroll
  for (int it = 0; it < GAT_N / 256; ++it) {
    int j = threadIdx.x + it * 256;
    float e = AxB[j] + ay;
    e = e > 0.f ? e : GAT_ALPHA * e;
    e = (arow[j] > 0) ? e : GAT_NEG_INF;
    ev[it] = e;
    lmax = fmaxf(lmax, e);
  }
  __shared__ float red[256];
  red[threadIdx.x] = lmax; __syncthreads();
  #pragma unroll
  for (int s = 128; s >= 1; s >>= 1) {
    if (threadIdx.x < s) red[threadIdx.x] = fmaxf(red[threadIdx.x], red[threadIdx.x + s]);
    __syncthreads();
  }
  const float m = red[0];
  __syncthreads();
  float lsum = 0.f;
  #pragma unroll
  for (int it = 0; it < GAT_N / 256; ++it) lsum += __expf(ev[it] - m);
  red[threadIdx.x] = lsum; __syncthreads();
  #pragma unroll
  for (int s = 128; s >= 1; s >>= 1) {
    if (threadIdx.x < s) red[threadIdx.x] += red[threadIdx.x + s];
    __syncthreads();
  }
  if (threadIdx.x == 0) { Mm[rowG] = m; Ss[rowG] = red[0]; }
}

// ---------- kernel 3b: featT[b][d][n] = (f16) feat[b][n][d]  (one-time transpose) ----------
__global__ void __launch_bounds__(256) k_t16(const float* __restrict__ feat,
                                             _Float16* __restrict__ featT) {
  __shared__ _Float16 tile[32][33];
  const int tx = threadIdx.x & 31, ty = threadIdx.x >> 5;
  const int nb = blockIdx.x * 32, db = blockIdx.y * 32, b = blockIdx.z;
  #pragma unroll
  for (int i = 0; i < 4; ++i) {
    int n = nb + ty + i * 8;
    tile[ty + i * 8][tx] = (_Float16)feat[((size_t)b * GAT_N + n) * GAT_D + db + tx];
  }
  __syncthreads();
  #pragma unroll
  for (int i = 0; i < 4; ++i) {
    int d = db + ty + i * 8;
    featT[((size_t)b * GAT_D + d) * GAT_N + nb + tx] = tile[tx][ty + i * 8];
  }
}

// ---------- kernel 3c: WT16[d][k] = (f16) W[k][d]  (one-time transpose) ----------
__global__ void __launch_bounds__(256) k_wt16(const float* __restrict__ W,
                                              _Float16* __restrict__ WT16) {
  __shared__ _Float16 tile[32][33];
  const int tx = threadIdx.x & 31, ty = threadIdx.x >> 5;
  const int kb = blockIdx.x * 32, db = blockIdx.y * 32;
  #pragma unroll
  for (int i = 0; i < 4; ++i)
    tile[ty + i * 8][tx] = (_Float16)W[(size_t)(kb + ty + i * 8) * GAT_D + db + tx];
  __syncthreads();
  #pragma unroll
  for (int i = 0; i < 4; ++i)
    WT16[(size_t)(db + ty + i * 8) * GAT_D + kb + tx] = tile[tx][ty + i * 8];
}

// ---------- kernel 4: h_prime = softmax(P) @ feat via v_wmma_f32_16x16x32_f16 ----------
// block = 256 threads (8 waves): 16 output rows x 256 dims; wave w owns dims [32w, 32w+32)
// Double-buffered: feat tile (async global->LDS from pre-transposed f16 featT) and P tile
// for iteration jt+1 are produced while tile jt is consumed by WMMA. One barrier / iter.
__global__ void __launch_bounds__(256) k_attn(
    const _Float16* __restrict__ featT, const int* __restrict__ adj,
    const float* __restrict__ Ax, const float* __restrict__ Ay,
    const float* __restrict__ Mm, const float* __restrict__ Ss,
    float* __restrict__ hp) {
  const int b  = blockIdx.y;
  const int i0 = blockIdx.x * 16;
  const int t = threadIdx.x;
  const int lane = t & 31;
  const int wave = t >> 5;

  __shared__ alignas(16) _Float16 fTT[2][GAT_D][40];  // [buf][d][k] B-tile, rows 80B
  __shared__ alignas(16) _Float16 pF[2][32][24];      // [buf] A tile in fragment layout
  __shared__ float ayS[16], mS[16], invS[16];

  if (t < 16) {
    int rg = b * GAT_N + i0 + t;
    ayS[t] = Ay[rg]; mS[t] = Mm[rg]; invS[t] = 1.0f / Ss[rg];
  }
  __syncthreads();

  const unsigned ldsF[2] = { (unsigned)(size_t)&fTT[0][0][0],
                             (unsigned)(size_t)&fTT[1][0][0] };
  const unsigned long long featTb =
      (unsigned long long)(size_t)(featT + (size_t)b * GAT_D * GAT_N);

  const float* AxB  = Ax + (size_t)b * GAT_N;
  const int*   adjB = adj + (size_t)b * GAT_N * GAT_N;

  // ---- A(P)-fragment generation mapping: thread t owns (lane Lp, elems h0,h0+1) ----
  const int Lp = t >> 3;
  const int h0 = (t & 7) * 2;
  const int rp = Lp & 15;                                  // matrix row M
  const int koffp = (Lp < 16) ? 0 : 16;
  const int K0 = h0 + koffp;
  const int* adjRow = adjB + (size_t)(i0 + rp) * GAT_N;
  const float ayv = ayS[rp], mv = mS[rp];

  // ---- per-thread async-copy geometry (4 x b128 per tile) ----
  const int dcp  = t >> 2;            // wait... set below per q
  (void)dcp;

  // ---- WMMA fragment geometry (ISA 7.12.2, wave32) ----
  const int ncol  = lane & 15;
  const int kbase = (lane < 16) ? 0 : 16;
  const int d0 = wave * 32;
  const int mbase = (lane < 16) ? 0 : 8;

  v8f acc0 = {}; v8f acc1 = {};

  // ---------- prologue: stage tile 0 into buffer 0 ----------
  #pragma unroll
  for (int q = 0; q < 4; ++q) {
    int ci = q * 256 + t;
    int d = ci >> 2, part = ci & 3;
    unsigned voff = (unsigned)((d * GAT_N + part * 8) * 2);
    unsigned loff = ldsF[0] + (unsigned)(d * 80 + part * 16);
    asm volatile("global_load_async_to_lds_b128 %0, %1, %2"
                 :: "v"(loff), "v"(voff), "s"(featTb) : "memory");
  }
  {
    int2   av  = *(const int2*)&adjRow[K0];
    float2 axv = *(const float2*)&AxB[K0];
    float e0 = axv.x + ayv; e0 = e0 > 0.f ? e0 : GAT_ALPHA * e0;
    float e1 = axv.y + ayv; e1 = e1 > 0.f ? e1 : GAT_ALPHA * e1;
    e0 = (av.x > 0) ? e0 : GAT_NEG_INF;
    e1 = (av.y > 0) ? e1 : GAT_NEG_INF;
    pF[0][Lp][h0]     = (_Float16)__expf(e0 - mv);
    pF[0][Lp][h0 + 1] = (_Float16)__expf(e1 - mv);
  }

  // ---------- main loop: one barrier per K-tile ----------
  for (int jt = 0; jt < GAT_N / 32; ++jt) {
    const int cb = jt & 1, nb = cb ^ 1;
    asm volatile("s_wait_asynccnt 0x0" ::: "memory");
    __syncthreads();

    const int jn = (jt + 1) * 32;
    if (jn < GAT_N) {
      // stage tile jt+1 (async copy + P compute) into the other buffer
      #pragma unroll
      for (int q = 0; q < 4; ++q) {
        int ci = q * 256 + t;
        int d = ci >> 2, part = ci & 3;
        unsigned voff = (unsigned)((d * GAT_N + jn + part * 8) * 2);
        unsigned loff = ldsF[nb] + (unsigned)(d * 80 + part * 16);
        asm volatile("global_load_async_to_lds_b128 %0, %1, %2"
                     :: "v"(loff), "v"(voff), "s"(featTb) : "memory");
      }
      int2   av  = *(const int2*)&adjRow[jn + K0];
      float2 axv = *(const float2*)&AxB[jn + K0];
      float e0 = axv.x + ayv; e0 = e0 > 0.f ? e0 : GAT_ALPHA * e0;
      float e1 = axv.y + ayv; e1 = e1 > 0.f ? e1 : GAT_ALPHA * e1;
      e0 = (av.x > 0) ? e0 : GAT_NEG_INF;
      e1 = (av.y > 0) ? e1 : GAT_NEG_INF;
      pF[nb][Lp][h0]     = (_Float16)__expf(e0 - mv);
      pF[nb][Lp][h0 + 1] = (_Float16)__expf(e1 - mv);
      if (jn + 32 < GAT_N)
        __builtin_prefetch(&adjRow[jn + 32 + K0], 0, 0);
    }

    // consume tile jt
    v16h Af  = *(const v16h*)&pF[cb][lane][0];
    v16h Bf0 = *(const v16h*)&fTT[cb][d0 + ncol][kbase];
    v16h Bf1 = *(const v16h*)&fTT[cb][d0 + 16 + ncol][kbase];

    acc0 = __builtin_amdgcn_wmma_f32_16x16x32_f16(false, Af, false, Bf0, (short)0, acc0, false, false);
    acc1 = __builtin_amdgcn_wmma_f32_16x16x32_f16(false, Af, false, Bf1, (short)0, acc1, false, false);
  }

  // scale by 1/s and store h_prime (C layout: VGPR r -> row r or r+8; lane%16 -> col)
  #pragma unroll
  for (int r = 0; r < 8; ++r) {
    int Mi = r + mbase;
    float sc = invS[Mi];
    size_t o = ((size_t)(b * GAT_N) + i0 + Mi) * GAT_D;
    hp[o + d0 + ncol]      = acc0[r] * sc;
    hp[o + d0 + 16 + ncol] = acc1[r] * sc;
  }
}

// ---------- kernel 5: out = elu(h_prime @ W) via WMMA ----------
// W fully LDS-resident (async copied once); all 8 A-fragment tiles prebuilt -> the
// K-loop is pure ds_load_b128 + WMMA with no barriers.
__global__ void __launch_bounds__(256) k_out(const float* __restrict__ hp,
                                             const _Float16* __restrict__ WT16,
                                             float* __restrict__ out) {
  const int row0 = blockIdx.x * 16;          // over B*N rows
  const int t = threadIdx.x, lane = t & 31, wave = t >> 5;

  __shared__ alignas(16) _Float16 wTT[GAT_D][264];  // full W (d-major), rows 528B
  __shared__ alignas(16) _Float16 aF[8][32][24];    // all K-tiles of A, fragment layout

  const unsigned ldsW = (unsigned)(size_t)&wTT[0][0];
  const unsigned long long wp = (unsigned long long)(size_t)WT16;

  // async-copy all of WT16 (128 KB payload) into LDS once: 32 x b128 per thread
  #pragma unroll
  for (int q = 0; q < 32; ++q) {
    int ci = q * 256 + t;
    int d = ci >> 5, part = ci & 31;
    unsigned voff = (unsigned)((d * GAT_D + part * 8) * 2);
    unsigned loff = ldsW + (unsigned)(d * 528 + part * 16);
    asm volatile("global_load_async_to_lds_b128 %0, %1, %2"
                 :: "v"(loff), "v"(voff), "s"(wp) : "memory");
  }

  const int Lp = t >> 3;
  const int h0 = (t & 7) * 2;
  const int rp = Lp & 15;
  const int koffp = (Lp < 16) ? 0 : 16;
  const int K0 = h0 + koffp;

  const int ncol  = lane & 15;
  const int kbase = (lane < 16) ? 0 : 16;
  const int d0 = wave * 32;
  const int mbase = (lane < 16) ? 0 : 8;

  // prebuild all 8 A-fragment tiles while W streams in
  #pragma unroll
  for (int kt = 0; kt < 8; ++kt) {
    float2 hv = *(const float2*)&hp[(size_t)(row0 + rp) * GAT_D + kt * 32 + K0];
    aF[kt][Lp][h0]     = (_Float16)hv.x;
    aF[kt][Lp][h0 + 1] = (_Float16)hv.y;
  }

  v8f acc0 = {}; v8f acc1 = {};

  asm volatile("s_wait_asynccnt 0x0" ::: "memory");
  __syncthreads();

  #pragma unroll
  for (int kt = 0; kt < 8; ++kt) {
    v16h Af  = *(const v16h*)&aF[kt][lane][0];
    v16h Bf0 = *(const v16h*)&wTT[d0 + ncol][kt * 32 + kbase];
    v16h Bf1 = *(const v16h*)&wTT[d0 + 16 + ncol][kt * 32 + kbase];

    acc0 = __builtin_amdgcn_wmma_f32_16x16x32_f16(false, Af, false, Bf0, (short)0, acc0, false, false);
    acc1 = __builtin_amdgcn_wmma_f32_16x16x32_f16(false, Af, false, Bf1, (short)0, acc1, false, false);
  }

  #pragma unroll
  for (int r = 0; r < 8; ++r) {
    float x0 = acc0[r], x1 = acc1[r];
    x0 = x0 > 0.f ? x0 : (__expf(x0) - 1.f);   // ELU, alpha=1
    x1 = x1 > 0.f ? x1 : (__expf(x1) - 1.f);
    size_t o = (size_t)(row0 + r + mbase) * GAT_D;
    out[o + d0 + ncol]      = x0;
    out[o + d0 + 16 + ncol] = x1;
  }
}

extern "C" void kernel_launch(void* const* d_in, const int* in_sizes, int n_in,
                              void* d_out, int out_size, void* d_ws, size_t ws_size,
                              hipStream_t stream) {
  const float* feat = (const float*)d_in[0];   // [B,N,D]
  const int*   adj  = (const int*)d_in[1];     // [B,N,N]
  const float* W    = (const float*)d_in[2];   // [D,D]
  const float* a1   = (const float*)d_in[3];   // [D,1]
  const float* a2   = (const float*)d_in[4];   // [D,1]
  float* out = (float*)d_out;                  // [B,N,D] f32

  const int BN = GAT_B * GAT_N;
  float* ws  = (float*)d_ws;
  float* wa1 = ws;
  float* wa2 = wa1 + GAT_D;
  float* Ax  = wa2 + GAT_D;
  float* Ay  = Ax + BN;
  float* Mm  = Ay + BN;
  float* Ss  = Mm + BN;
  float* hp  = Ss + BN;                            // [B,N,D] f32 scratch
  _Float16* featT = (_Float16*)(hp + (size_t)BN * GAT_D);   // [B,D,N] f16
  _Float16* WT16  = featT + (size_t)GAT_B * GAT_D * GAT_N;  // [D,D] f16 (d-major)

  k_wa   <<<1, GAT_D, 0, stream>>>(W, a1, a2, wa1, wa2);
  k_axay <<<BN / 8, 256, 0, stream>>>(feat, wa1, wa2, Ax, Ay);
  k_stats<<<BN, 256, 0, stream>>>(adj, Ax, Ay, Mm, Ss);
  k_t16  <<<dim3(GAT_N / 32, GAT_D / 32, GAT_B), 256, 0, stream>>>(feat, featT);
  k_wt16 <<<dim3(GAT_D / 32, GAT_D / 32), 256, 0, stream>>>(W, WT16);
  k_attn <<<dim3(GAT_N / 16, GAT_B), 256, 0, stream>>>(featT, adj, Ax, Ay, Mm, Ss, hp);
  k_out  <<<BN / 16, 256, 0, stream>>>(hp, WT16, out);
}
